// EncoderDGLConcat_55559696941459
// MI455X (gfx1250) — matrix operand order repaired and checked by reading, not compiled, not run on hardware
//
#include <hip/hip_runtime.h>
#include <hip/hip_bf16.h>

// ---------------------------------------------------------------------------
// GraphCast-style grid->mesh encoder, fused for MI455X (gfx1250, wave32).
// bf16 WMMA (V_WMMA_F32_16X16X32_BF16), f32 accumulation.
//   kernel 0: transpose+convert weights fp32 -> bf16  (Wt[n][k])
//   kernel 1 (MODE 0): edge MLP  [E,1536]->[E,512], LN, atomic segment-sum
//   kernel 2 (MODE 1): dst MLP   [M,1024]->[M,512], LN, residual add
//   kernel 3 (MODE 2): src MLP   [G, 512]->[G,512], LN, residual add
// Block = 256 threads = 8 waves; tile = 32 rows x 512 cols.
// Each wave: 64-col slice, 2 M-tiles x 4 N-tiles (B fragment reused twice).
// ---------------------------------------------------------------------------

#define N_GRID 131072
#define N_MESH 40962
#define N_EDGE 131072
#define DN     512
#define MT     32        // rows per block

typedef __attribute__((ext_vector_type(16))) __bf16 v16bf;
typedef __attribute__((ext_vector_type(8)))  float  v8f;

union BF16x16 { uint4 q[2]; v16bf v; };

static __device__ __forceinline__ unsigned short f2bf(float f) {
    unsigned int u = __float_as_uint(f);
    unsigned int r = (u + 0x7FFFu + ((u >> 16) & 1u)) >> 16;   // RNE
    return (unsigned short)r;
}

// A-fragment (16x32 bf16) from LDS, row-major A[.][K].
// ISA 7.12.2: lane half h: elems 0..7 = K[k0+8h..+7], elems 8..15 = K[k0+16+8h..+7]
static __device__ __forceinline__ v16bf load_a_frag(const unsigned short* arow, int k0, int h) {
    BF16x16 u;
    u.q[0] = *(const uint4*)(arow + k0 + 8 * h);
    u.q[1] = *(const uint4*)(arow + k0 + 16 + 8 * h);
    return u.v;
}

// B-fragment (32x16 bf16) from transposed weights Wt[n][k]:
// lane = column, half h: elems 0..7 = K[k0+8h..+7], elems 8..15 = K[k0+16+8h..+7].
static __device__ __forceinline__ v16bf load_b_frag(const unsigned short* wcol, int k0, int h) {
    BF16x16 u;
    u.q[0] = *(const uint4*)(wcol + k0 + 8 * h);
    u.q[1] = *(const uint4*)(wcol + k0 + 16 + 8 * h);
    return u.v;
}

// Wave-level GEMM: C[32 x 64] += A[32 x K] * W[K x 64] for this wave's slice.
template<int K>
static __device__ __forceinline__ void wave_gemm(const unsigned short* ldsA,
                                                 const unsigned short* wt,
                                                 int n0, int r, int h, v8f acc[2][4]) {
    const unsigned short* arow0 = ldsA + r * K;
    const unsigned short* arow1 = ldsA + (16 + r) * K;
    for (int k0 = 0; k0 < K; k0 += 32) {
        v16bf a0 = load_a_frag(arow0, k0, h);
        v16bf a1 = load_a_frag(arow1, k0, h);
#pragma unroll
        for (int t = 0; t < 4; ++t) {
            const unsigned short* wcol = wt + (size_t)(n0 + t * 16 + r) * K;
            if (k0 + 32 < K) __builtin_prefetch(wcol + k0 + 32, 0, 3);  // global_prefetch_b8 (near)
            v16bf b = load_b_frag(wcol, k0, h);
            acc[0][t] = __builtin_amdgcn_wmma_f32_16x16x32_bf16(
                false, a0, false, b, (short)0, acc[0][t], false, false);
            acc[1][t] = __builtin_amdgcn_wmma_f32_16x16x32_bf16(
                false, a1, false, b, (short)0, acc[1][t], false, false);
        }
    }
}

// ---------------------------------------------------------------------------
// Weight prep: Wt[n*K + k] = bf16(W[k*N + n])
// ---------------------------------------------------------------------------
__global__ void transpose_bf16_kernel(const float* __restrict__ w,
                                      unsigned short* __restrict__ wt,
                                      int K, int N) {
    int idx = blockIdx.x * 256 + threadIdx.x;
    if (idx >= K * N) return;
    int n = idx / K;
    int k = idx - n * K;
    wt[idx] = f2bf(w[(size_t)k * N + n]);
}

// ---------------------------------------------------------------------------
// Fused MLP: stage -> GEMM1 -> SiLU -> GEMM2 -> LayerNorm -> epilogue
// MODE 0: edge  (gather grid||mesh||efeat, scatter atomicAdd into agg)
// MODE 1: dst   (agg||mesh, residual mesh, store to out)
// MODE 2: src   (grid, residual grid, store to out)
// LDS: region0 = max(A-tile bf16, out-tile f32) [aliased], then hidden bf16.
// ---------------------------------------------------------------------------
template<int K1, int MODE>
__global__ __launch_bounds__(256) void mlp_fused_kernel(
    const float* __restrict__ srcA, const float* __restrict__ srcB,
    const float* __restrict__ srcC,
    const int* __restrict__ src_idx, const int* __restrict__ dst_idx,
    const unsigned short* __restrict__ w1t, const float* __restrict__ b1,
    const unsigned short* __restrict__ w2t, const float* __restrict__ b2,
    const float* __restrict__ gamma, const float* __restrict__ beta,
    const float* __restrict__ resid,
    float* __restrict__ out, float* __restrict__ agg, int nrows) {

    constexpr size_t R0 = ((size_t)MT * K1 * 2 > (size_t)MT * DN * 4)
                              ? (size_t)MT * K1 * 2 : (size_t)MT * DN * 4;

    extern __shared__ char smem[];
    unsigned short* lds_a = (unsigned short*)smem;            // MT x K1 bf16
    float*          lds_o = (float*)smem;                     // MT x 512 f32 (aliases lds_a)
    unsigned short* lds_h = (unsigned short*)(smem + R0);     // MT x 512 bf16
    int* s_src = (int*)(smem + R0 + (size_t)MT * DN * 2);
    int* s_dst = s_src + MT;

    const int tid  = threadIdx.x;
    const int lane = tid & 31;
    const int wave = tid >> 5;
    const int r    = lane & 15;    // row-in-tile / col-in-tile
    const int h    = lane >> 4;    // lane half
    const int n0   = wave * 64;    // this wave's output-column base
    const int row0 = blockIdx.x * MT;

    if (MODE == 0) {
        if (tid < MT) {
            s_src[tid] = src_idx[row0 + tid];
            s_dst[tid] = dst_idx[row0 + tid];
        }
        __syncthreads();
    }

    // ---- stage input tile (fp32 gather -> bf16 in LDS) ------------------
    constexpr int CH = MT * K1 / 4;                 // float4 chunks
    for (int i = tid; i < CH; i += 256) {
        int rr = i / (K1 / 4);
        int k  = (i - rr * (K1 / 4)) * 4;
        float4 f = make_float4(0.f, 0.f, 0.f, 0.f);
        if (MODE == 0) {
            if (k < DN)            f = *(const float4*)(srcA + (size_t)s_src[rr] * DN + k);
            else if (k < 2 * DN)   f = *(const float4*)(srcB + (size_t)s_dst[rr] * DN + (k - DN));
            else                   f = *(const float4*)(srcC + (size_t)(row0 + rr) * DN + (k - 2 * DN));
        } else if (MODE == 1) {
            int rg = row0 + rr;
            if (rg < nrows) {
                if (k < DN) f = *(const float4*)(srcA + (size_t)rg * DN + k);
                else        f = *(const float4*)(srcB + (size_t)rg * DN + (k - DN));
            }
        } else {
            int rg = row0 + rr;
            if (rg < nrows) f = *(const float4*)(srcA + (size_t)rg * DN + k);
        }
        ushort4 b;
        b.x = f2bf(f.x); b.y = f2bf(f.y); b.z = f2bf(f.z); b.w = f2bf(f.w);
        *(ushort4*)(lds_a + rr * K1 + k) = b;
    }
    __syncthreads();

    // ---- GEMM1: [32 x K1] @ [K1 x 512] ----------------------------------
    v8f acc[2][4] = {};
    wave_gemm<K1>(lds_a, w1t, n0, r, h, acc);

    // bias + SiLU -> bf16 hidden tile
#pragma unroll
    for (int t = 0; t < 4; ++t) {
        int c = n0 + t * 16 + r;
        float bias = b1[c];
#pragma unroll
        for (int mi = 0; mi < 2; ++mi) {
#pragma unroll
            for (int v = 0; v < 8; ++v) {
                int m = 16 * mi + 8 * h + v;
                float x = acc[mi][t][v] + bias;
                float s = x / (1.f + __expf(-x));
                lds_h[m * DN + c] = f2bf(s);
            }
        }
    }
    __syncthreads();   // hidden complete; all GEMM1 reads of lds_a also complete

    // ---- GEMM2: [32 x 512] @ [512 x 512] --------------------------------
    v8f acc2[2][4] = {};
    wave_gemm<DN>(lds_h, w2t, n0, r, h, acc2);

    // bias -> f32 staging (aliases lds_a, which is dead now)
#pragma unroll
    for (int t = 0; t < 4; ++t) {
        int c = n0 + t * 16 + r;
        float bias = b2[c];
#pragma unroll
        for (int mi = 0; mi < 2; ++mi) {
#pragma unroll
            for (int v = 0; v < 8; ++v) {
                int m = 16 * mi + 8 * h + v;
                lds_o[m * DN + c] = acc2[mi][t][v] + bias;
            }
        }
    }
    __syncthreads();

    // ---- LayerNorm over D=512 per row, then epilogue --------------------
    // wave w owns rows 4w..4w+3; lane l covers cols l + 32*j
#pragma unroll
    for (int rr = 0; rr < 4; ++rr) {
        int row = wave * 4 + rr;
        float sum = 0.f, ssq = 0.f;
#pragma unroll
        for (int j = 0; j < 16; ++j) {
            float x = lds_o[row * DN + lane + 32 * j];
            sum += x;
            ssq += x * x;
        }
#pragma unroll
        for (int o = 16; o > 0; o >>= 1) {
            sum += __shfl_xor(sum, o, 32);
            ssq += __shfl_xor(ssq, o, 32);
        }
        float mu  = sum * (1.f / (float)DN);
        float var = ssq * (1.f / (float)DN) - mu * mu;
        float rs  = rsqrtf(var + 1e-5f);

        int rg = row0 + row;
#pragma unroll
        for (int j = 0; j < 16; ++j) {
            int c = lane + 32 * j;
            float val = (lds_o[row * DN + c] - mu) * rs * gamma[c] + beta[c];
            if (MODE == 0) {
                unsafeAtomicAdd(&agg[(size_t)s_dst[row] * DN + c], val);
            } else {
                if (rg < nrows)
                    out[(size_t)rg * DN + c] = resid[(size_t)rg * DN + c] + val;
            }
        }
    }
}

// ---------------------------------------------------------------------------
extern "C" void kernel_launch(void* const* d_in, const int* in_sizes, int n_in,
                              void* d_out, int out_size, void* d_ws, size_t ws_size,
                              hipStream_t stream) {
    (void)in_sizes; (void)n_in; (void)out_size; (void)ws_size;

    const float* g2m     = (const float*)d_in[0];
    const float* grid    = (const float*)d_in[1];
    const float* mesh    = (const float*)d_in[2];
    const float* edge_w1 = (const float*)d_in[3];
    const float* edge_b1 = (const float*)d_in[4];
    const float* edge_w2 = (const float*)d_in[5];
    const float* edge_b2 = (const float*)d_in[6];
    const float* edge_g  = (const float*)d_in[7];
    const float* edge_be = (const float*)d_in[8];
    const float* dst_w1  = (const float*)d_in[9];
    const float* dst_b1  = (const float*)d_in[10];
    const float* dst_w2  = (const float*)d_in[11];
    const float* dst_b2  = (const float*)d_in[12];
    const float* dst_g   = (const float*)d_in[13];
    const float* dst_be  = (const float*)d_in[14];
    const float* src_w1  = (const float*)d_in[15];
    const float* src_b1  = (const float*)d_in[16];
    const float* src_w2  = (const float*)d_in[17];
    const float* src_b2  = (const float*)d_in[18];
    const float* src_g   = (const float*)d_in[19];
    const float* src_be  = (const float*)d_in[20];
    const int*   src_idx = (const int*)d_in[21];
    const int*   dst_idx = (const int*)d_in[22];

    float* out_grid = (float*)d_out;                              // [N_GRID, D]
    float* out_mesh = (float*)d_out + (size_t)N_GRID * DN;        // [N_MESH, D]

    char* ws = (char*)d_ws;
    size_t off = 0;
    auto take = [&](size_t bytes) { char* p = ws + off; off += (bytes + 255) & ~(size_t)255; return p; };
    unsigned short* ew1t = (unsigned short*)take((size_t)DN * 3 * DN * 2);
    unsigned short* ew2t = (unsigned short*)take((size_t)DN * DN * 2);
    unsigned short* dw1t = (unsigned short*)take((size_t)DN * 2 * DN * 2);
    unsigned short* dw2t = (unsigned short*)take((size_t)DN * DN * 2);
    unsigned short* sw1t = (unsigned short*)take((size_t)DN * DN * 2);
    unsigned short* sw2t = (unsigned short*)take((size_t)DN * DN * 2);
    float*          agg  = (float*)take((size_t)N_MESH * DN * 4);

    auto tconv = [&](const float* w, unsigned short* wt, int K, int N) {
        int total = K * N;
        transpose_bf16_kernel<<<(total + 255) / 256, 256, 0, stream>>>(w, wt, K, N);
    };
    tconv(edge_w1, ew1t, 3 * DN, DN);
    tconv(edge_w2, ew2t, DN, DN);
    tconv(dst_w1,  dw1t, 2 * DN, DN);
    tconv(dst_w2,  dw2t, DN, DN);
    tconv(src_w1,  sw1t, DN, DN);
    tconv(src_w2,  sw2t, DN, DN);

    hipMemsetAsync(agg, 0, (size_t)N_MESH * DN * 4, stream);

    auto smem_bytes = [](int K1) {
        size_t r0 = (size_t)MT * K1 * 2;
        size_t o  = (size_t)MT * DN * 4;
        if (o > r0) r0 = o;
        return r0 + (size_t)MT * DN * 2 + 2 * MT * sizeof(int);
    };

    // Edge MLP + segment-sum scatter
    mlp_fused_kernel<3 * DN, 0><<<N_EDGE / MT, 256, smem_bytes(3 * DN), stream>>>(
        grid, mesh, g2m, src_idx, dst_idx,
        ew1t, edge_b1, ew2t, edge_b2, edge_g, edge_be,
        nullptr, nullptr, agg, N_EDGE);

    // Dst (mesh) MLP with residual
    mlp_fused_kernel<2 * DN, 1><<<(N_MESH + MT - 1) / MT, 256, smem_bytes(2 * DN), stream>>>(
        agg, mesh, nullptr, nullptr, nullptr,
        dw1t, dst_b1, dw2t, dst_b2, dst_g, dst_be,
        mesh, out_mesh, nullptr, N_MESH);

    // Src (grid) MLP with residual
    mlp_fused_kernel<DN, 2><<<N_GRID / MT, 256, smem_bytes(DN), stream>>>(
        grid, nullptr, nullptr, nullptr, nullptr,
        sw1t, src_b1, sw2t, src_b2, src_g, src_be,
        grid, out_grid, nullptr, N_GRID);
}
